// _TGATLayer_70557722738860
// MI455X (gfx1250) — compile-verified, matrix operand order checked by reading, not run on hardware
//
#include <hip/hip_runtime.h>
#include <math.h>

// ---------------- types ----------------
typedef __bf16 bf16;
typedef __attribute__((ext_vector_type(16))) __bf16 v16bf;
typedef __attribute__((ext_vector_type(8)))  float  v8f;

#define DN    256      // D_IN == D_OUT
#define KNBR  10
#define HEADS 8
#define HD    32
#define FT    100
#define FTP   128      // padded time dim
#define RSQRT_HD 0.17677669529663687f   // 1/sqrt(32)

// round-to-nearest-even f32 -> bf16
static __device__ __forceinline__ bf16 f2bf(float f) {
  union { float f; unsigned u; } c; c.f = f;
  unsigned r = c.u + 0x7fffu + ((c.u >> 16) & 1u);
  unsigned short h = (unsigned short)(r >> 16);
  return __builtin_bit_cast(bf16, h);
}
static __device__ __forceinline__ float bf2f(bf16 b) {
  unsigned short h = __builtin_bit_cast(unsigned short, b);
  union { float f; unsigned u; } c; c.u = ((unsigned)h) << 16;
  return c.f;
}

// Load one 16x32 bf16 WMMA A/B fragment from a row-major array.
// ISA 7.12.2 16-bit A layout: lanes 0-15 hold M=lane, K=c0+{0..7,16..23};
// lanes 16-31 hold M=lane-16, K=c0+{8..15,24..31}. B uses the mirrored
// convention (lane = output column, reads along K of W[d][*]).
static __device__ __forceinline__ v16bf load_frag(const bf16* base, int stride,
                                                  int row0, int c0) {
  int lane = threadIdx.x & 31;
  const bf16* p = base + (size_t)(row0 + (lane & 15)) * stride
                       + c0 + ((lane >> 4) << 3);
  v16bf f;
#pragma unroll
  for (int i = 0; i < 8; ++i) { f[i] = p[i]; f[i + 8] = p[i + 16]; }
  return f;
}

static __device__ __forceinline__ v8f wmma_bf16(v16bf a, v16bf b, v8f c) {
  return __builtin_amdgcn_wmma_f32_16x16x32_bf16(false, a, false, b,
                                                 (short)0, c, false, false);
}

// ---------------- kernel 1: fold edge context into output bias ----------------
__global__ void k_edge(const float* __restrict__ We, const float* __restrict__ be,
                       const float* __restrict__ ef, const float* __restrict__ Wc,
                       const float* __restrict__ bc, float* __restrict__ cbias) {
  __shared__ float sE[DN];
  int d = threadIdx.x;
  float a = be[d];
  for (int e = 0; e < 172; ++e) a += We[d * 172 + e] * ef[e];
  sE[d] = a;
  __syncthreads();
  float cb = bc[d];
  for (int j = 0; j < DN; ++j) cb += Wc[d * 768 + 256 + j] * sE[j];
  cbias[d] = cb;
}

// ---------------- kernel 2: convert / repack weights to bf16 ----------------
// [0,65536)   Wq   [65536,131072) Wk   [131072,196608) Wv
// [196608,229376) Wt zero-padded 100->128
// [229376,360448) Wc2 = [Wc[:,0:256] | Wc[:,512:768]]
__global__ void k_convert(const float* __restrict__ Wq, const float* __restrict__ Wk,
                          const float* __restrict__ Wv, const float* __restrict__ Wt,
                          const float* __restrict__ Wc,
                          bf16* __restrict__ wq, bf16* __restrict__ wk,
                          bf16* __restrict__ wv, bf16* __restrict__ wt,
                          bf16* __restrict__ wc2) {
  int idx = blockIdx.x * blockDim.x + threadIdx.x;
  if (idx < 65536) {
    wq[idx] = f2bf(Wq[idx]);
  } else if (idx < 131072) {
    int j = idx - 65536;  wk[j] = f2bf(Wk[j]);
  } else if (idx < 196608) {
    int j = idx - 131072; wv[j] = f2bf(Wv[j]);
  } else if (idx < 229376) {
    int j = idx - 196608; int d = j >> 7, c = j & 127;
    wt[j] = f2bf(c < FT ? Wt[d * FT + c] : 0.0f);
  } else if (idx < 360448) {
    int j = idx - 229376; int d = j >> 9, c = j & 511;
    wc2[j] = f2bf(c < 256 ? Wc[d * 768 + c] : Wc[d * 768 + 256 + c]);
  }
}

// ---------------- kernel 3: q = X @ Wq^T + bq (WMMA) ----------------
__global__ __launch_bounds__(256) void k_qproj(const float* __restrict__ X,
                                               const bf16* __restrict__ wq,
                                               const float* __restrict__ bq,
                                               float* __restrict__ q) {
  __shared__ bf16 sX[16][264];
  int r0 = blockIdx.x * 16;
  int t = threadIdx.x;
  for (int e = t; e < 16 * DN; e += 256) {
    int r = e >> 8, c = e & 255;
    sX[r][c] = f2bf(X[(size_t)(r0 + r) * DN + c]);
  }
  __syncthreads();
  int lane = t & 31, w = t >> 5, half = lane >> 4;
#pragma unroll
  for (int cti = 0; cti < 2; ++cti) {
    int ct = w * 2 + cti;
    int col = ct * 16 + (lane & 15);
    v8f acc = {};
#pragma unroll
    for (int kk = 0; kk < 8; ++kk)
      acc = wmma_bf16(load_frag(&sX[0][0], 264, 0, kk * 32),
                      load_frag(wq, DN, ct * 16, kk * 32), acc);
    float bias = bq[col];
#pragma unroll
    for (int v = 0; v < 8; ++v) {
      int row = v + half * 8;
      q[(size_t)(r0 + row) * DN + col] = acc[v] + bias;
    }
  }
}

// ---------------- kernel 4: fused neighbor K/V/time projection + attention ----
// One WG = 8 nodes = 80 (n,k) rows = 5 WMMA row tiles. Keys are never stored:
// each key C-tile is dotted with q via shfl_xor butterfly + ds_add_f32 scores.
__global__ __launch_bounds__(256) void k_neigh(
    const float* __restrict__ X, const int* __restrict__ nidx,
    const float* __restrict__ ntime, const float* __restrict__ wtime,
    const float* __restrict__ btime, const float* __restrict__ q,
    const bf16* __restrict__ wk, const bf16* __restrict__ wv,
    const bf16* __restrict__ wt, const float* __restrict__ bk,
    const float* __restrict__ bv, const float* __restrict__ bt,
    float* __restrict__ attn) {
  __shared__ bf16  sXg[80][264];      // gathered neighbor rows (bf16)
  __shared__ bf16  sT[80][136];       // cos time encoding, padded to 128
  __shared__ bf16  sV[80][264];       // value projections
  __shared__ float sQ[8][DN];         // q rows for this WG's nodes
  __shared__ float sSc[8][HEADS][16]; // scores [node][head][k]
  __shared__ int   sNid[80];
  __shared__ float sTim[80];

  int n0 = blockIdx.x * 8;
  int t = threadIdx.x;
  if (t < 80) {
    int i = t / KNBR, k = t - i * KNBR;
    sNid[t] = nidx[(size_t)(n0 + i) * KNBR + k];
    sTim[t] = ntime[(size_t)(n0 + i) * KNBR + k];
  }
  { float* sp = &sSc[0][0][0];
    for (int e = t; e < 8 * HEADS * 16; e += 256) sp[e] = 0.0f; }
  __syncthreads();
  for (int e = t; e < 80 * DN; e += 256) {
    int r = e >> 8, c = e & 255;
    sXg[r][c] = f2bf(X[(size_t)sNid[r] * DN + c]);
  }
  for (int e = t; e < 8 * DN; e += 256) {
    int i = e >> 8, c = e & 255;
    sQ[i][c] = q[(size_t)(n0 + i) * DN + c];
  }
  for (int e = t; e < 80 * FTP; e += 256) {
    int r = e >> 7, f = e & 127;
    float v = 0.0f;
    if (f < FT) v = cosf(sTim[r] * wtime[f] + btime[f]);
    sT[r][f] = f2bf(v);
  }
  __syncthreads();

  int lane = t & 31, w = t >> 5, half = lane >> 4;
  int cl = lane & 15;

  // ---- keys -> scores (bias folded, scaled by 1/sqrt(HD)) ----
#pragma unroll
  for (int cti = 0; cti < 2; ++cti) {
    int ct = w * 2 + cti;
    int h = ct >> 1;                 // 16 cols of a tile live in one head
    int col = ct * 16 + cl;
    v16bf B[8];
#pragma unroll
    for (int kk = 0; kk < 8; ++kk) B[kk] = load_frag(wk, DN, ct * 16, kk * 32);
    float bias = bk[col];
    for (int rt = 0; rt < 5; ++rt) {
      v8f acc = {};
#pragma unroll
      for (int kk = 0; kk < 8; ++kk)
        acc = wmma_bf16(load_frag(&sXg[0][0], 264, rt * 16, kk * 32), B[kk], acc);
#pragma unroll
      for (int v = 0; v < 8; ++v) {
        int row = rt * 16 + v + half * 8;     // same for all 16 lanes of a half
        int node = row / KNBR, kn = row - node * KNBR;
        float val = (acc[v] + bias) * sQ[node][col] * RSQRT_HD;
        val += __shfl_xor(val, 1, 32);
        val += __shfl_xor(val, 2, 32);
        val += __shfl_xor(val, 4, 32);
        val += __shfl_xor(val, 8, 32);
        if (cl == 0) atomicAdd(&sSc[node][h][kn], val);   // ds_add_f32
      }
    }
  }
  // ---- time context mean -> scores (weight 1/HD) ----
#pragma unroll
  for (int cti = 0; cti < 2; ++cti) {
    int ct = w * 2 + cti;
    int h = ct >> 1;
    int col = ct * 16 + cl;
    v16bf B[4];
#pragma unroll
    for (int kk = 0; kk < 4; ++kk) B[kk] = load_frag(wt, FTP, ct * 16, kk * 32);
    float bias = bt[col];
    for (int rt = 0; rt < 5; ++rt) {
      v8f acc = {};
#pragma unroll
      for (int kk = 0; kk < 4; ++kk)
        acc = wmma_bf16(load_frag(&sT[0][0], 136, rt * 16, kk * 32), B[kk], acc);
#pragma unroll
      for (int v = 0; v < 8; ++v) {
        int row = rt * 16 + v + half * 8;
        int node = row / KNBR, kn = row - node * KNBR;
        float val = (acc[v] + bias) * (1.0f / 32.0f);
        val += __shfl_xor(val, 1, 32);
        val += __shfl_xor(val, 2, 32);
        val += __shfl_xor(val, 4, 32);
        val += __shfl_xor(val, 8, 32);
        if (cl == 0) atomicAdd(&sSc[node][h][kn], val);
      }
    }
  }
  // ---- values -> LDS ----
#pragma unroll
  for (int cti = 0; cti < 2; ++cti) {
    int ct = w * 2 + cti;
    int col = ct * 16 + cl;
    v16bf B[8];
#pragma unroll
    for (int kk = 0; kk < 8; ++kk) B[kk] = load_frag(wv, DN, ct * 16, kk * 32);
    float bias = bv[col];
    for (int rt = 0; rt < 5; ++rt) {
      v8f acc = {};
#pragma unroll
      for (int kk = 0; kk < 8; ++kk)
        acc = wmma_bf16(load_frag(&sXg[0][0], 264, rt * 16, kk * 32), B[kk], acc);
#pragma unroll
      for (int v = 0; v < 8; ++v) {
        int row = rt * 16 + v + half * 8;
        sV[row][col] = f2bf(acc[v] + bias);
      }
    }
  }
  __syncthreads();

  // ---- softmax over K=10 per (node, head), in place ----
  if (t < 64) {
    int i = t >> 3, h = t & 7;
    float mx = -1e30f;
    for (int k = 0; k < KNBR; ++k) mx = fmaxf(mx, sSc[i][h][k]);
    float s = 0.0f;
    for (int k = 0; k < KNBR; ++k) {
      float e = expf(sSc[i][h][k] - mx);
      sSc[i][h][k] = e; s += e;
    }
    float inv = 1.0f / s;
    for (int k = 0; k < KNBR; ++k) sSc[i][h][k] *= inv;
  }
  __syncthreads();

  // ---- weighted sum over neighbors ----
  int i = t >> 5, dd = t & 31;
#pragma unroll
  for (int h = 0; h < HEADS; ++h) {
    int d = h * 32 + dd;
    float a = 0.0f;
    for (int k = 0; k < KNBR; ++k)
      a += sSc[i][h][k] * bf2f(sV[i * KNBR + k][d]);
    attn[(size_t)(n0 + i) * DN + d] = a;
  }
}

// ---------------- kernel 5: [attn|q] @ Wc2^T + cbias, +X, LayerNorm ----------
__global__ __launch_bounds__(256) void k_final(
    const float* __restrict__ attn, const float* __restrict__ q,
    const bf16* __restrict__ wc2, const float* __restrict__ cbias,
    const float* __restrict__ X, const float* __restrict__ g,
    const float* __restrict__ b, float* __restrict__ out) {
  __shared__ bf16  sA[16][520];
  __shared__ float sC[16][264];
  __shared__ float sRed[16][16];
  __shared__ float sMu[16], sRstd[16], sFlag[16];

  int r0 = blockIdx.x * 16;
  int t = threadIdx.x;
  for (int e = t; e < 16 * 512; e += 256) {
    int r = e >> 9, c = e & 511;
    float v = (c < 256) ? attn[(size_t)(r0 + r) * DN + c]
                        : q[(size_t)(r0 + r) * DN + (c - 256)];
    sA[r][c] = f2bf(v);
  }
  __syncthreads();

  int lane = t & 31, w = t >> 5, half = lane >> 4;
#pragma unroll
  for (int cti = 0; cti < 2; ++cti) {
    int ct = w * 2 + cti;
    int col = ct * 16 + (lane & 15);
    v8f acc = {};
#pragma unroll
    for (int kk = 0; kk < 16; ++kk)
      acc = wmma_bf16(load_frag(&sA[0][0], 520, 0, kk * 32),
                      load_frag(wc2, 512, ct * 16, kk * 32), acc);
    float cb = cbias[col];
#pragma unroll
    for (int v = 0; v < 8; ++v) sC[v + half * 8][col] = acc[v] + cb;
  }
  __syncthreads();

  // residual + LayerNorm; 16 threads per row
  int row = t >> 4, p = t & 15;
  float s = 0.0f;
  for (int j = 0; j < 16; ++j) {
    int c = p * 16 + j;
    float y = sC[row][c] + X[(size_t)(r0 + row) * DN + c];
    sC[row][c] = y; s += y;
  }
  sRed[row][p] = s;
  __syncthreads();
  if (p == 0) {
    float m = 0.0f;
    for (int j = 0; j < 16; ++j) m += sRed[row][j];
    sMu[row] = m * (1.0f / DN);
  }
  __syncthreads();
  float mu = sMu[row];
  s = 0.0f;
  for (int j = 0; j < 16; ++j) {
    int c = p * 16 + j;
    float d = sC[row][c] - mu; s += d * d;
  }
  sRed[row][p] = s;
  __syncthreads();
  if (p == 0) {
    float v = 0.0f;
    for (int j = 0; j < 16; ++j) v += sRed[row][j];
    sRstd[row] = rsqrtf(v * (1.0f / DN) + 1e-5f);
  }
  __syncthreads();
  float rstd = sRstd[row];
  s = 0.0f;
  for (int j = 0; j < 16; ++j) {
    int c = p * 16 + j;
    float o = (sC[row][c] - mu) * rstd * g[c] + b[c];
    out[(size_t)(r0 + row) * DN + c] = o;
    s += fabsf(o);
  }
  sRed[row][p] = s;
  __syncthreads();
  if (p == 0) {
    float a = 0.0f;
    for (int j = 0; j < 16; ++j) a += sRed[row][j];
    sFlag[row] = a;
  }
  __syncthreads();
  if (sFlag[row] == 0.0f) {  // all-zero-row fallback (matches reference)
    for (int j = 0; j < 16; ++j) {
      int c = p * 16 + j;
      out[(size_t)(r0 + row) * DN + c] = X[(size_t)(r0 + row) * DN + c];
    }
  }
}

// ---------------- launcher ----------------
extern "C" void kernel_launch(void* const* d_in, const int* in_sizes, int n_in,
                              void* d_out, int out_size, void* d_ws, size_t ws_size,
                              hipStream_t stream) {
  const float* X     = (const float*)d_in[0];
  const float* ntime = (const float*)d_in[1];
  const float* ef    = (const float*)d_in[2];
  const float* Wq    = (const float*)d_in[3];
  const float* bq    = (const float*)d_in[4];
  const float* Wk    = (const float*)d_in[5];
  const float* bk    = (const float*)d_in[6];
  const float* Wv    = (const float*)d_in[7];
  const float* bv    = (const float*)d_in[8];
  const float* We    = (const float*)d_in[9];
  const float* be    = (const float*)d_in[10];
  const float* Wt    = (const float*)d_in[11];
  const float* bt    = (const float*)d_in[12];
  const float* Wc    = (const float*)d_in[13];
  const float* bc    = (const float*)d_in[14];
  // d_in[15] Wo, d_in[16] bo: unused by the reference
  const float* ln_g  = (const float*)d_in[17];
  const float* ln_b  = (const float*)d_in[18];
  const float* wtime = (const float*)d_in[19];
  const float* btime = (const float*)d_in[20];
  const int*   nidx  = (const int*)d_in[21];

  const int N = in_sizes[0] / DN;  // 32768
  char* ws = (char*)d_ws;
  size_t off = 0;
  float* q     = (float*)(ws + off); off += (size_t)N * DN * 4;
  float* attn  = (float*)(ws + off); off += (size_t)N * DN * 4;
  bf16* wq_bf  = (bf16*)(ws + off);  off += 65536 * 2;
  bf16* wk_bf  = (bf16*)(ws + off);  off += 65536 * 2;
  bf16* wv_bf  = (bf16*)(ws + off);  off += 65536 * 2;
  bf16* wt_bf  = (bf16*)(ws + off);  off += 32768 * 2;
  bf16* wc2_bf = (bf16*)(ws + off);  off += 131072 * 2;
  float* cbias = (float*)(ws + off); off += 256 * 4;
  (void)ws_size; (void)n_in; (void)out_size;

  k_edge<<<1, 256, 0, stream>>>(We, be, ef, Wc, bc, cbias);
  k_convert<<<(360448 + 255) / 256, 256, 0, stream>>>(Wq, Wk, Wv, Wt, Wc,
                                                      wq_bf, wk_bf, wv_bf,
                                                      wt_bf, wc2_bf);
  k_qproj<<<N / 16, 256, 0, stream>>>(X, wq_bf, bq, q);
  k_neigh<<<N / 8, 256, 0, stream>>>(X, nidx, ntime, wtime, btime, q,
                                     wk_bf, wv_bf, wt_bf, bk, bv, bt, attn);
  k_final<<<N / 16, 256, 0, stream>>>(attn, q, wc2_bf, cbias, X, ln_g, ln_b,
                                      (float*)d_out);
}
